// GatedDeltaRule_91079076479244
// MI455X (gfx1250) — compile-verified, matrix operand order checked by reading, not run on hardware
//
#include <hip/hip_runtime.h>
#include <hip/hip_bf16.h>

// ---------------------------------------------------------------------------
// Gated delta-rule layer for MI455X (gfx1250, wave32, WMMA).
// B=8, T=2048, D=1024, C=128.
//   q,k,v,g = x@W + b   (bf16 WMMA GEMMs, f32 accum, async-LDS double buffer)
//   alpha,beta = sigmoid(x@Wa/Wb + b)
//   serial scan over T with S (128x128 f32) held in registers (1 WG / batch)
//   y = o@Wo + bo       (bf16 WMMA GEMM)
// ---------------------------------------------------------------------------

#define NB 8
#define NT 2048
#define ND 1024
#define NC 128
#define NBT (NB * NT)

typedef __bf16 bf16x16 __attribute__((ext_vector_type(16)));
typedef float floatx8 __attribute__((ext_vector_type(8)));
typedef int vint4 __attribute__((vector_size(16)));

union FragU {
    bf16x16 v;
    uint4 q[2];
};

// ---- gfx1250 async global->LDS path (guarded; falls back to sync copy) -----
#if defined(__has_builtin)
#if __has_builtin(__builtin_amdgcn_global_load_async_to_lds_b128)
#define USE_ASYNC_LDS 1
#endif
#endif

#ifdef USE_ASYNC_LDS
#define AS_GLOBAL __attribute__((address_space(1)))
#define AS_LDS __attribute__((address_space(3)))
__device__ __forceinline__ void async_cp16(const unsigned short* g,
                                           unsigned short* l) {
    __builtin_amdgcn_global_load_async_to_lds_b128((AS_GLOBAL vint4*)g,
                                                   (AS_LDS vint4*)l, 0, 0);
}
__device__ __forceinline__ void wait_async0() {
#if __has_builtin(__builtin_amdgcn_s_wait_asynccnt)
    __builtin_amdgcn_s_wait_asynccnt(0);
#else
    asm volatile("s_wait_asynccnt 0x0" ::: "memory");
#endif
}
#endif

__device__ __forceinline__ unsigned short f32_to_bf16(float f) {
    union { float f; unsigned u; } v;
    v.f = f;
    unsigned r = v.u + 0x7FFFu + ((v.u >> 16) & 1u);  // round-to-nearest-even
    return (unsigned short)(r >> 16);
}

__device__ __forceinline__ float sigmoidf_dev(float x) {
    return 1.0f / (1.0f + __expf(-x));
}

// ------------------------- f32 -> bf16 convert ------------------------------
__global__ __launch_bounds__(256) void cvt_bf16_kernel(
    const float* __restrict__ src, unsigned short* __restrict__ dst, int total) {
    int base = (blockIdx.x * blockDim.x + threadIdx.x) * 4;
    if (base + 3 < total) {
        float4 v = *(const float4*)(src + base);
        dst[base + 0] = f32_to_bf16(v.x);
        dst[base + 1] = f32_to_bf16(v.y);
        dst[base + 2] = f32_to_bf16(v.z);
        dst[base + 3] = f32_to_bf16(v.w);
    }
}

// --------------------- transpose (K,N) f32 -> (N,K) bf16 --------------------
__global__ __launch_bounds__(256) void transpose_bf16_kernel(
    const float* __restrict__ W, unsigned short* __restrict__ Wt, int K, int N) {
    int idx = blockIdx.x * blockDim.x + threadIdx.x;
    if (idx < K * N) {
        int n = idx / K;
        int k = idx - n * K;
        Wt[(size_t)n * K + k] = f32_to_bf16(W[(size_t)k * N + n]);
    }
}

// ------------------------------- bf16 GEMM ----------------------------------
// C(M,N) = A(M,K) * Bt(N,K)^T + bias ; epi: 0 = none, 1 = sigmoid.
// 128x128 tile per block, 256 threads = 8 waves, each wave a 16x128 strip.
// Double-buffered LDS tiles; stage of chunk c+1 (async global->LDS when
// available) overlaps the 8 WMMAs of chunk c.
__global__ __launch_bounds__(256) void gemm_bf16_kernel(
    const unsigned short* __restrict__ A,   // (M,K) row-major bf16
    const unsigned short* __restrict__ Bt,  // (N,K) row-major bf16
    const float* __restrict__ bias,         // (N)
    float* __restrict__ out,                // (M,N) f32
    int M, int N, int K, int epi) {
    __shared__ __align__(16) unsigned short Abuf[2][128 * 48];  // 32 cols + pad
    __shared__ __align__(16) unsigned short Bbuf[2][128 * 48];

    const int tid  = threadIdx.x;
    const int wave = tid >> 5;
    const int lane = tid & 31;
    const int m0 = blockIdx.x * 128;
    const int n0 = blockIdx.y * 128;

    floatx8 acc[8];
#pragma unroll
    for (int i = 0; i < 8; ++i)
#pragma unroll
        for (int j = 0; j < 8; ++j) acc[i][j] = 0.0f;

    const int lr   = lane & 15;
    const int koff = (lane >> 4) * 8;

    auto stage = [&](int buf, int kc) {
#pragma unroll
        for (int i = 0; i < 2; ++i) {
            int lin = tid + i * 256;  // 0..511 = 128 rows x 4 uint4/row
            int row = lin >> 2;
            int c4  = lin & 3;
            const unsigned short* ga = A + (size_t)(m0 + row) * K + kc + c4 * 8;
            const unsigned short* gb = Bt + (size_t)(n0 + row) * K + kc + c4 * 8;
            unsigned short* la = &Abuf[buf][row * 48 + c4 * 8];
            unsigned short* lb = &Bbuf[buf][row * 48 + c4 * 8];
#ifdef USE_ASYNC_LDS
            async_cp16(ga, la);
            async_cp16(gb, lb);
#else
            *(uint4*)la = *(const uint4*)ga;
            *(uint4*)lb = *(const uint4*)gb;
#endif
        }
    };

    const int nchunks = K >> 5;  // K / 32
    stage(0, 0);
#ifdef USE_ASYNC_LDS
    wait_async0();
#endif
    __syncthreads();

    for (int c = 0; c < nchunks; ++c) {
        const int cur = c & 1;
        if (c + 1 < nchunks) stage(cur ^ 1, (c + 1) * 32);

        FragU af;
        af.q[0] = *(const uint4*)&Abuf[cur][(wave * 16 + lr) * 48 + koff];
        af.q[1] = *(const uint4*)&Abuf[cur][(wave * 16 + lr) * 48 + koff + 16];
#pragma unroll
        for (int nt = 0; nt < 8; ++nt) {
            FragU bfr;
            bfr.q[0] = *(const uint4*)&Bbuf[cur][(nt * 16 + lr) * 48 + koff];
            bfr.q[1] = *(const uint4*)&Bbuf[cur][(nt * 16 + lr) * 48 + koff + 16];
            acc[nt] = __builtin_amdgcn_wmma_f32_16x16x32_bf16(
                false, af.v, false, bfr.v, (short)0, acc[nt], false, false);
        }
#ifdef USE_ASYNC_LDS
        wait_async0();  // our async writes to the next buffer are done
#endif
        __syncthreads();  // everyone's writes visible before buffer swap
    }

    // Epilogue: C/D layout — VGPR v, lane<16: (M=v, N=lane); lane>=16: (M=v+8).
    const int hi = lane >> 4;
#pragma unroll
    for (int nt = 0; nt < 8; ++nt) {
        int col  = n0 + nt * 16 + lr;
        float bv = bias[col];
#pragma unroll
        for (int v = 0; v < 8; ++v) {
            int mrow  = m0 + wave * 16 + v + hi * 8;
            float val = acc[nt][v] + bv;
            if (epi == 1) val = sigmoidf_dev(val);
            out[(size_t)mrow * N + col] = val;
        }
    }
}

// --------------------------- alpha / beta -----------------------------------
__global__ __launch_bounds__(128) void ab_kernel(
    const float* __restrict__ x, const float* __restrict__ Wa,
    const float* __restrict__ ba, const float* __restrict__ Wb,
    const float* __restrict__ bb, float* __restrict__ alpha,
    float* __restrict__ beta) {
    __shared__ float ra[128], rb[128];
    int row = blockIdx.x;
    int tid = threadIdx.x;
    const float* xr = x + (size_t)row * ND;
    float pa = 0.f, pb = 0.f;
#pragma unroll
    for (int i = 0; i < 8; ++i) {
        int j = tid + i * 128;
        float xv = xr[j];
        pa += xv * Wa[j];
        pb += xv * Wb[j];
    }
    ra[tid] = pa;
    rb[tid] = pb;
    __syncthreads();
    for (int s = 64; s > 0; s >>= 1) {
        if (tid < s) { ra[tid] += ra[tid + s]; rb[tid] += rb[tid + s]; }
        __syncthreads();
    }
    if (tid == 0) {
        alpha[row] = sigmoidf_dev(ra[0] + ba[0]);
        beta[row]  = sigmoidf_dev(rb[0] + bb[0]);
    }
}

// ------------------------------- scan ---------------------------------------
// One block per batch; 256 threads; thread (r,h)=(tid>>1,tid&1) owns
// S[r][h*64 .. h*64+63] in registers.
__global__ __launch_bounds__(256) void scan_kernel(
    const float* __restrict__ qf, const float* __restrict__ kf,
    const float* __restrict__ vf, const float* __restrict__ gf,
    const float* __restrict__ alpha, const float* __restrict__ beta,
    unsigned short* __restrict__ ob /* bf16 (B,T,C) */) {
    __shared__ float qs[128], ks[128], vs[128], gs[128];
    __shared__ float red1[64];
    __shared__ float plds[256 * 33];
    __shared__ float red2[64 * 4];
    __shared__ float sc[4];

    const int tid = threadIdx.x;
    const int b   = blockIdx.x;
    const int r   = tid >> 1;
    const int h   = tid & 1;

    float S[64];
#pragma unroll
    for (int j = 0; j < 64; ++j) S[j] = 0.0f;

    const size_t base = (size_t)b * NT * NC;

    for (int t = 0; t < NT; ++t) {
        const size_t rb_ = base + (size_t)t * NC;
        if (tid < 128) {
            qs[tid] = qf[rb_ + tid];
            ks[tid] = kf[rb_ + tid];
        } else {
            int j = tid - 128;
            vs[j] = vf[rb_ + j];
            gs[j] = gf[rb_ + j];
        }
        __syncthreads();

        if (tid < 64)
            red1[tid] = ks[tid] * ks[tid] + ks[tid + 64] * ks[tid + 64];
        __syncthreads();
        if (tid == 0) {
            float ss = 0.f;
            for (int i = 0; i < 64; ++i) ss += red1[i];
            float inv = 1.0f / fmaxf(sqrtf(ss), 1e-12f);
            size_t rowIdx = (size_t)b * NT + t;
            sc[0] = alpha[rowIdx];
            sc[1] = beta[rowIdx];
            sc[2] = inv;
        }
        __syncthreads();
        const float a   = sc[0];
        const float bt_ = sc[1];
        const float inv = sc[2];

        // k row half into registers (L2-normalized).
        float kreg[64];
        const float4* kp = (const float4*)&ks[h * 64];
#pragma unroll
        for (int i = 0; i < 16; ++i) {
            float4 kv = kp[i];
            kreg[i * 4 + 0] = kv.x * inv;
            kreg[i * 4 + 1] = kv.y * inv;
            kreg[i * 4 + 2] = kv.z * inv;
            kreg[i * 4 + 3] = kv.w * inv;
        }

        // m = S k  (in-thread over 64 cols + pair combine across halves)
        float pm = 0.f;
#pragma unroll
        for (int j = 0; j < 64; ++j) pm += S[j] * kreg[j];
        pm += __shfl_xor(pm, 1, 32);

        const float u  = bt_ * vs[r] - a * bt_ * pm;
        const float qr = qs[r];

        // Rank-1 state update: S = a*S + u * k^T
#pragma unroll
        for (int j = 0; j < 64; ++j) S[j] = a * S[j] + u * kreg[j];

        // o[c] = sum_r q[r] * S[r][c], times gate; LDS transpose-reduce,
        // two passes of 32 columns per half.
        for (int pass = 0; pass < 2; ++pass) {
            float* pp = &plds[tid * 33];
#pragma unroll
            for (int jj = 0; jj < 32; ++jj) pp[jj] = qr * S[pass * 32 + jj];
            __syncthreads();

            int col  = tid >> 2;   // 0..63 -> (hh, jj)
            int part = tid & 3;
            int hh   = col >> 5;
            int jj   = col & 31;
            float s2 = 0.f;
            for (int rr = part * 32; rr < part * 32 + 32; ++rr)
                s2 += plds[(rr * 2 + hh) * 33 + jj];
            red2[col * 4 + part] = s2;
            __syncthreads();

            if (tid < 64) {
                int c = (tid < 32) ? (pass * 32 + tid)
                                   : (64 + pass * 32 + (tid - 32));
                float o = red2[tid * 4] + red2[tid * 4 + 1] +
                          red2[tid * 4 + 2] + red2[tid * 4 + 3];
                o *= gs[c];
                ob[rb_ + c] = f32_to_bf16(o);
            }
            __syncthreads();
        }
    }
}

// ------------------------------- launch -------------------------------------
extern "C" void kernel_launch(void* const* d_in, const int* in_sizes, int n_in,
                              void* d_out, int out_size, void* d_ws,
                              size_t ws_size, hipStream_t stream) {
    const float* x  = (const float*)d_in[0];
    const float* Wq = (const float*)d_in[1];
    const float* bq = (const float*)d_in[2];
    const float* Wk = (const float*)d_in[3];
    const float* bk = (const float*)d_in[4];
    const float* Wv = (const float*)d_in[5];
    const float* bv = (const float*)d_in[6];
    const float* Wa = (const float*)d_in[7];
    const float* ba = (const float*)d_in[8];
    const float* Wb = (const float*)d_in[9];
    const float* bb = (const float*)d_in[10];
    const float* Wg = (const float*)d_in[11];
    const float* bg = (const float*)d_in[12];
    const float* Wo = (const float*)d_in[13];
    const float* bo = (const float*)d_in[14];

    char* ws = (char*)d_ws;
    size_t off = 0;
    auto take = [&](size_t bytes) -> char* {
        char* p = ws + off;
        off = (off + bytes + 255) & ~(size_t)255;
        return p;
    };

    unsigned short* xb  = (unsigned short*)take((size_t)NBT * ND * 2);
    unsigned short* wtq = (unsigned short*)take((size_t)ND * NC * 2);
    unsigned short* wtk = (unsigned short*)take((size_t)ND * NC * 2);
    unsigned short* wtv = (unsigned short*)take((size_t)ND * NC * 2);
    unsigned short* wtg = (unsigned short*)take((size_t)ND * NC * 2);
    unsigned short* wot = (unsigned short*)take((size_t)NC * ND * 2);
    float* qf = (float*)take((size_t)NBT * NC * 4);
    float* kf = (float*)take((size_t)NBT * NC * 4);
    float* vf = (float*)take((size_t)NBT * NC * 4);
    float* gf = (float*)take((size_t)NBT * NC * 4);
    float* al = (float*)take((size_t)NBT * 4);
    float* be = (float*)take((size_t)NBT * 4);
    unsigned short* ob = (unsigned short*)take((size_t)NBT * NC * 2);

    // 1) x -> bf16
    {
        int total = NBT * ND;
        cvt_bf16_kernel<<<total / 1024, 256, 0, stream>>>(x, xb, total);
    }
    // 2) weights -> (N,K) bf16
    {
        int blocks = (ND * NC + 255) / 256;
        transpose_bf16_kernel<<<blocks, 256, 0, stream>>>(Wq, wtq, ND, NC);
        transpose_bf16_kernel<<<blocks, 256, 0, stream>>>(Wk, wtk, ND, NC);
        transpose_bf16_kernel<<<blocks, 256, 0, stream>>>(Wv, wtv, ND, NC);
        transpose_bf16_kernel<<<blocks, 256, 0, stream>>>(Wg, wtg, ND, NC);
        transpose_bf16_kernel<<<blocks, 256, 0, stream>>>(Wo, wot, NC, ND);
    }
    // 3) projections (WMMA bf16)
    {
        dim3 gp(NBT / 128, NC / 128);
        gemm_bf16_kernel<<<gp, 256, 0, stream>>>(xb, wtq, bq, qf, NBT, NC, ND, 0);
        gemm_bf16_kernel<<<gp, 256, 0, stream>>>(xb, wtk, bk, kf, NBT, NC, ND, 0);
        gemm_bf16_kernel<<<gp, 256, 0, stream>>>(xb, wtv, bv, vf, NBT, NC, ND, 0);
        gemm_bf16_kernel<<<gp, 256, 0, stream>>>(xb, wtg, bg, gf, NBT, NC, ND, 1);
    }
    // 4) alpha/beta
    ab_kernel<<<NBT, 128, 0, stream>>>(x, Wa, ba, Wb, bb, al, be);
    // 5) serial scan, one workgroup per batch, state in registers
    scan_kernel<<<NB, 256, 0, stream>>>(qf, kf, vf, gf, al, be, ob);
    // 6) output projection (WMMA bf16)
    {
        dim3 gy(NBT / 128, ND / 128);
        gemm_bf16_kernel<<<gy, 256, 0, stream>>>(ob, wot, bo, (float*)d_out,
                                                 NBT, ND, NC, 0);
    }
}
